// MHA_15719580303901
// MI455X (gfx1250) — compile-verified, matrix operand order checked by reading, not run on hardware
//
#include <hip/hip_runtime.h>
#include <hip/hip_bf16.h>

typedef __attribute__((ext_vector_type(16))) _Float16 v16h;
typedef __attribute__((ext_vector_type(8)))  float    v8f;

#define T_SEQ 4096
#define C_EMB 1024
#define NH    16
#define HS    64
#define WIN   256
#define SINK  4

#define WMMA_F16(A_,B_,C_) __builtin_amdgcn_wmma_f32_16x16x32_f16(false, (A_), false, (B_), (short)0, (C_), false, false)

// Load a 16-half A/B fragment for v_wmma_f32_16x16x32_f16.
// Per CDNA5 ISA layout: lane (lo = lane&15) holds row/col `lo`; half-wave (hi)
// selects K sub-blocks: elements e=0..7 -> K = hi*8 + e, e=8..15 -> K = 16 + hi*8 + (e-8).
// Both chunks are 8 contiguous halves = one 16B load each.
__device__ __forceinline__ v16h load_frag(const _Float16* __restrict__ row, int hi8) {
    union { v16h v; float4 f[2]; } u;
    u.f[0] = *reinterpret_cast<const float4*>(row + hi8);
    u.f[1] = *reinterpret_cast<const float4*>(row + 16 + hi8);
    return u.v;
}

// ---------------- f32 -> f16 conversion ----------------
__global__ __launch_bounds__(256) void cvt_f32_f16(const float* __restrict__ s,
                                                   _Float16* __restrict__ d, int n) {
    int i = 4 * (blockIdx.x * blockDim.x + threadIdx.x);
    if (i < n) {
        float4 v = *reinterpret_cast<const float4*>(s + i);
        union { _Float16 h[4]; float2 f2; } u;
        u.h[0] = (_Float16)v.x; u.h[1] = (_Float16)v.y;
        u.h[2] = (_Float16)v.z; u.h[3] = (_Float16)v.w;
        *reinterpret_cast<float2*>(d + i) = u.f2;
    }
}

// ---------------- GEMM: out = A(4096x1024) @ W^T(1024x1024) + bias ----------------
// One wave computes a 32x64 output strip (2 M-subtiles x 4 N-subtiles); the two
// M-subtiles share the B-fragments, so per k-step: 2 A-loads + 4 B-loads feed 8 WMMAs.
// MODE 0: f16 row-major out; MODE 1: f16 transposed (out[o*T_SEQ+t], for V^T);
// MODE 2: f32 row-major (final output).
template <int MODE>
__global__ __launch_bounds__(256) void gemm32x64(const _Float16* __restrict__ A,
                                                 const _Float16* __restrict__ W,
                                                 const float* __restrict__ bias,
                                                 void* __restrict__ out) {
    const int K = C_EMB;
    int wv   = (blockIdx.x * blockDim.x + threadIdx.x) >> 5;
    int lane = threadIdx.x & 31;
    int lo   = lane & 15;
    int hi8  = (lane >> 4) * 8;
    const int nstrips = C_EMB / 64;              // 16
    int t0 = (wv / nstrips) * 32;
    int o0 = (wv % nstrips) * 64;

    const _Float16* arow0 = A + (size_t)(t0 + lo) * K;
    const _Float16* arow1 = A + (size_t)(t0 + 16 + lo) * K;
    v8f acc[2][4] = {};
    for (int k0 = 0; k0 < K; k0 += 32) {
        v16h a0 = load_frag(arow0 + k0, hi8);
        v16h a1 = load_frag(arow1 + k0, hi8);
#pragma unroll
        for (int nb = 0; nb < 4; nb++) {
            v16h b = load_frag(W + (size_t)(o0 + nb * 16 + lo) * K + k0, hi8);
            acc[0][nb] = WMMA_F16(a0, b, acc[0][nb]);
            acc[1][nb] = WMMA_F16(a1, b, acc[1][nb]);
        }
    }
#pragma unroll
    for (int mt = 0; mt < 2; mt++) {
        int tb = t0 + mt * 16;
#pragma unroll
        for (int nb = 0; nb < 4; nb++) {
            int o = o0 + nb * 16 + lo;
            float bv = bias[o];
            if (MODE == 1) {
                union { _Float16 h[8]; float4 f; } u;
#pragma unroll
                for (int r = 0; r < 8; r++) u.h[r] = (_Float16)(acc[mt][nb][r] + bv);
                *reinterpret_cast<float4*>((_Float16*)out + (size_t)o * T_SEQ + tb + hi8) = u.f;
            } else {
#pragma unroll
                for (int r = 0; r < 8; r++) {
                    int t = tb + r + hi8;
                    float v = acc[mt][nb][r] + bv;
                    if (MODE == 0) ((_Float16*)out)[(size_t)t * C_EMB + o] = (_Float16)v;
                    else           ((float*)out)[(size_t)t * C_EMB + o]    = v;
                }
            }
        }
    }
}

// ---------------- Flash attention, sliding window + sinks ----------------
// One wave per (head, 16-query tile). Iterates 32-key blocks: sink block + band.
__global__ __launch_bounds__(256) void attn_kernel(const _Float16* __restrict__ Q,
                                                   const _Float16* __restrict__ Km,
                                                   const _Float16* __restrict__ Vt,
                                                   _Float16* __restrict__ O) {
    __shared__ __align__(16) _Float16 plds[8][512];   // 16x32 f16 P tile per wave
    int wib  = threadIdx.x >> 5;
    int gw   = blockIdx.x * 8 + wib;
    int head = gw >> 8;
    int qt   = gw & 255;
    int q0   = qt * 16;
    int lane = threadIdx.x & 31;
    int lo   = lane & 15;
    int hi8  = (lane >> 4) * 8;
    _Float16* pl = plds[wib];

    v16h aq0 = load_frag(Q + (size_t)(q0 + lo) * C_EMB + head * HS,      hi8);
    v16h aq1 = load_frag(Q + (size_t)(q0 + lo) * C_EMB + head * HS + 32, hi8);

    float m_i[8], l_i[8];
#pragma unroll
    for (int r = 0; r < 8; r++) { m_i[r] = -1e30f; l_i[r] = 0.0f; }
    v8f acc[4] = {};

    int c_start = (q0 > (WIN - 1)) ? ((q0 - (WIN - 1)) & ~31) : 0;
    int c_last  = q0 + 15;
    int nblk = (c_last - c_start) / 32 + 1 + (c_start > 0 ? 1 : 0);

    for (int step = 0; step < nblk; step++) {
        int c0 = (c_start > 0) ? ((step == 0) ? 0 : c_start + (step - 1) * 32)
                               : step * 32;
        // ---- S = Q K^T for two 16-key tiles ----
        v8f s0 = {}, s1 = {};
        {
            v16h b00 = load_frag(Km + (size_t)(c0 + lo) * C_EMB + head * HS,      hi8);
            v16h b01 = load_frag(Km + (size_t)(c0 + lo) * C_EMB + head * HS + 32, hi8);
            s0 = WMMA_F16(aq0, b00, s0);
            s0 = WMMA_F16(aq1, b01, s0);
            v16h b10 = load_frag(Km + (size_t)(c0 + 16 + lo) * C_EMB + head * HS,      hi8);
            v16h b11 = load_frag(Km + (size_t)(c0 + 16 + lo) * C_EMB + head * HS + 32, hi8);
            s1 = WMMA_F16(aq0, b10, s1);
            s1 = WMMA_F16(aq1, b11, s1);
        }
        // ---- mask + scale (1/sqrt(64)) ----
        float rm[8];
#pragma unroll
        for (int r = 0; r < 8; r++) {
            int t  = q0 + r + hi8;
            int ca = c0 + lo;
            int cb = c0 + 16 + lo;
            float a0 = ((ca <= t) && ((ca + (WIN - 1) >= t) || (ca < SINK))) ? s0[r] * 0.125f : -1e30f;
            float a1 = ((cb <= t) && ((cb + (WIN - 1) >= t) || (cb < SINK))) ? s1[r] * 0.125f : -1e30f;
            s0[r] = a0; s1[r] = a1;
            rm[r] = fmaxf(a0, a1);
        }
        // ---- row max over 16 lanes (half-wave) ----
#pragma unroll
        for (int m = 8; m >= 1; m >>= 1)
#pragma unroll
            for (int r = 0; r < 8; r++) rm[r] = fmaxf(rm[r], __shfl_xor(rm[r], m, 32));
        // ---- online softmax update; write P tile (f16) to LDS ----
        float rs[8];
#pragma unroll
        for (int r = 0; r < 8; r++) {
            float mn = fmaxf(m_i[r], rm[r]);
            float sc = __expf(m_i[r] - mn);
            m_i[r] = mn;
            float p0 = __expf(s0[r] - mn);
            float p1 = __expf(s1[r] - mn);
            rs[r] = p0 + p1;
            l_i[r] *= sc;
#pragma unroll
            for (int nb = 0; nb < 4; nb++) acc[nb][r] *= sc;
            pl[(r + hi8) * 32 + lo]      = (_Float16)p0;
            pl[(r + hi8) * 32 + 16 + lo] = (_Float16)p1;
        }
#pragma unroll
        for (int m = 8; m >= 1; m >>= 1)
#pragma unroll
            for (int r = 0; r < 8; r++) rs[r] += __shfl_xor(rs[r], m, 32);
#pragma unroll
        for (int r = 0; r < 8; r++) l_i[r] += rs[r];
        // ---- re-layout P: D-layout -> A-fragment via LDS (same-wave RAW) ----
        asm volatile("s_wait_dscnt 0" ::: "memory");
        v16h ap = load_frag(pl + lo * 32, hi8);
        // ---- acc += P @ V  (V stored transposed: Vt[o][t], contiguous in key) ----
#pragma unroll
        for (int nb = 0; nb < 4; nb++) {
            v16h bv = load_frag(Vt + (size_t)(head * HS + nb * 16 + lo) * T_SEQ + c0, hi8);
            acc[nb] = WMMA_F16(ap, bv, acc[nb]);
        }
    }
    // ---- normalize and store f16 output [t][h*64+d] ----
#pragma unroll
    for (int r = 0; r < 8; r++) {
        float inv = 1.0f / l_i[r];
        int t = q0 + r + hi8;
#pragma unroll
        for (int nb = 0; nb < 4; nb++)
            O[(size_t)t * C_EMB + head * HS + nb * 16 + lo] = (_Float16)(acc[nb][r] * inv);
    }
}

extern "C" void kernel_launch(void* const* d_in, const int* in_sizes, int n_in,
                              void* d_out, int out_size, void* d_ws, size_t ws_size,
                              hipStream_t stream) {
    const float* x  = (const float*)d_in[0];
    const float* qw = (const float*)d_in[1];
    const float* qb = (const float*)d_in[2];
    const float* kw = (const float*)d_in[3];
    const float* kb = (const float*)d_in[4];
    const float* vw = (const float*)d_in[5];
    const float* vb = (const float*)d_in[6];
    const float* ow = (const float*)d_in[7];
    const float* ob = (const float*)d_in[8];

    const size_t NX = (size_t)T_SEQ * C_EMB;   // 4M elements
    const size_t NW = (size_t)C_EMB * C_EMB;   // 1M elements

    _Float16* ws  = (_Float16*)d_ws;
    _Float16* xf  = ws;            // x f16
    _Float16* qwf = xf  + NX;
    _Float16* kwf = qwf + NW;
    _Float16* vwf = kwf + NW;
    _Float16* owf = vwf + NW;
    _Float16* qf  = owf + NW;      // Q  [t][c]
    _Float16* kf  = qf  + NX;      // K  [t][c]
    _Float16* vtf = kf  + NX;      // V^T [c][t]
    _Float16* of  = vtf + NX;      // attention out f16 [t][c]

    // 1) convert to f16
    cvt_f32_f16<<<(int)(NX / 4 / 256), 256, 0, stream>>>(x,  xf,  (int)NX);
    cvt_f32_f16<<<(int)(NW / 4 / 256), 256, 0, stream>>>(qw, qwf, (int)NW);
    cvt_f32_f16<<<(int)(NW / 4 / 256), 256, 0, stream>>>(kw, kwf, (int)NW);
    cvt_f32_f16<<<(int)(NW / 4 / 256), 256, 0, stream>>>(vw, vwf, (int)NW);
    cvt_f32_f16<<<(int)(NW / 4 / 256), 256, 0, stream>>>(ow, owf, (int)NW);

    // 2) projections: 2048 waves = (4096/32)*(1024/64); 8 waves/block -> 256 blocks
    gemm32x64<0><<<256, 256, 0, stream>>>(xf, qwf, qb, (void*)qf);
    gemm32x64<0><<<256, 256, 0, stream>>>(xf, kwf, kb, (void*)kf);
    gemm32x64<1><<<256, 256, 0, stream>>>(xf, vwf, vb, (void*)vtf);

    // 3) attention: 16 heads * 256 query tiles = 4096 waves -> 512 blocks
    attn_kernel<<<512, 256, 0, stream>>>(qf, kf, vtf, of);

    // 4) output projection -> f32 d_out
    gemm32x64<2><<<256, 256, 0, stream>>>(of, owf, ob, d_out);
}
